// CNN_GRU_GNN_75917841924470
// MI455X (gfx1250) — compile-verified, matrix operand order checked by reading, not run on hardware
//
#include <hip/hip_runtime.h>
#include <hip/hip_bf16.h>
#include <math.h>

typedef __attribute__((ext_vector_type(16))) _Float16 v16h;
typedef __attribute__((ext_vector_type(8)))  _Float16 v8h;
typedef __attribute__((ext_vector_type(8)))  float    v8f;

#define NEG_SLOPE 0.2f
#define BN_EPS    1e-5f

static __device__ __forceinline__ v16h mk16(v8h lo, v8h hi) {
  v16h r;
#pragma unroll
  for (int i = 0; i < 8; ++i) { r[i] = lo[i]; r[i + 8] = hi[i]; }
  return r;
}

// ---------------------------------------------------------------------------
// WMMA GEMM: C[Mpad,Npad](f32) = A[Mpad,Kpad](f16) * Bt[Npad,Kpad](f16)^T
// One wave per 32(M) x 64(N) tile: 8 accumulators; 2 A fragments reused
// across 4 B fragments -> 12 b128 loads per 8 WMMAs per K-step.
// Requires: Mpad%32==0, Npad%64==0, Kpad%32==0; padded regions zero-filled.
// ---------------------------------------------------------------------------
__global__ void __launch_bounds__(32)
k_gemm_wmma(const _Float16* __restrict__ A, const _Float16* __restrict__ Bt,
            float* __restrict__ C, int Kpad, int Npad)
{
  const int lane = threadIdx.x & 31;
  const int hf   = lane >> 4;     // which K-half this lane holds
  const int r    = lane & 15;     // row (A) / col (B) within tile
  const long m0 = (long)blockIdx.y * 32;
  const long n0 = (long)blockIdx.x * 64;
  v8f acc[2][4];
#pragma unroll
  for (int i = 0; i < 2; ++i)
#pragma unroll
    for (int j = 0; j < 4; ++j) acc[i][j] = (v8f){};
  const _Float16* pa0 = A + (m0 + r) * (long)Kpad + hf * 8;
  const _Float16* pa1 = pa0 + 16 * (long)Kpad;
  const _Float16* pb[4];
#pragma unroll
  for (int j = 0; j < 4; ++j)
    pb[j] = Bt + (n0 + 16 * j + r) * (long)Kpad + hf * 8;
  for (int k = 0; k < Kpad; k += 32) {
    __builtin_prefetch((const void*)(pa0 + k + 32), 0, 1);
    __builtin_prefetch((const void*)(pa1 + k + 32), 0, 1);
    v16h a0 = mk16(*(const v8h*)(pa0 + k), *(const v8h*)(pa0 + k + 16));
    v16h a1 = mk16(*(const v8h*)(pa1 + k), *(const v8h*)(pa1 + k + 16));
#pragma unroll
    for (int j = 0; j < 4; ++j) {
      v16h b = mk16(*(const v8h*)(pb[j] + k), *(const v8h*)(pb[j] + k + 16));
      acc[0][j] = __builtin_amdgcn_wmma_f32_16x16x32_f16(false, a0, false, b, (short)0, acc[0][j], false, false);
      acc[1][j] = __builtin_amdgcn_wmma_f32_16x16x32_f16(false, a1, false, b, (short)0, acc[1][j], false, false);
    }
  }
  // C/D layout: VGPR v -> row tile_m + hf*8 + v, lane r -> col n0 + 16*j + r
#pragma unroll
  for (int i = 0; i < 2; ++i) {
    float* pc = C + (m0 + 16 * i + hf * 8) * (long)Npad + n0 + r;
#pragma unroll
    for (int v = 0; v < 8; ++v)
#pragma unroll
      for (int j = 0; j < 4; ++j)
        pc[(long)v * Npad + 16 * j] = acc[i][j][v];
  }
}

// ---------------------------------------------------------------------------
// im2col for 3x3x3 'same' conv. chFirst: src [BT,C,D,H,W]; else channels-last
// [BT*D*H*W, C]. dst f16 [Mpad, Kpad], zero-padded.
// ---------------------------------------------------------------------------
__global__ void k_im2col(const float* __restrict__ src, _Float16* __restrict__ dst,
                         int BT, int C, int D, int H, int W,
                         int Kreal, int Kpad, int Mpad, int chFirst)
{
  size_t idx = (size_t)blockIdx.x * blockDim.x + threadIdx.x;
  size_t total = (size_t)Mpad * Kpad;
  if (idx >= total) return;
  int k = (int)(idx % Kpad);
  long m = (long)(idx / Kpad);
  const long M = (long)BT * D * H * W;
  float v = 0.f;
  if (m < M && k < Kreal) {
    int w  = (int)(m % W);
    int h  = (int)((m / W) % H);
    int d  = (int)((m / ((long)W * H)) % D);
    int bt = (int)(m / ((long)W * H * D));
    int c  = k / 27, t = k % 27;
    int kd = t / 9, kh = (t / 3) % 3, kw = t % 3;
    int id = d + kd - 1, ih = h + kh - 1, iw = w + kw - 1;
    if (id >= 0 && id < D && ih >= 0 && ih < H && iw >= 0 && iw < W) {
      if (chFirst)
        v = src[((((long)bt * C + c) * D + id) * H + ih) * (long)W + iw];
      else
        v = src[((((long)bt * D + id) * H + ih) * (long)W + iw) * C + c];
    }
  }
  dst[idx] = (_Float16)v;
}

// weight fp32 [Nr,Kr] row-major -> f16 [Npad, Kpad], zero pad. (already [N,K])
__global__ void k_w16(const float* __restrict__ src, _Float16* __restrict__ dst,
                      int Nr, int Kr, int Kpad, int Npad)
{
  size_t idx = (size_t)blockIdx.x * blockDim.x + threadIdx.x;
  if (idx >= (size_t)Npad * Kpad) return;
  int k = (int)(idx % Kpad), n = (int)(idx / Kpad);
  float v = (n < Nr && k < Kr) ? src[(size_t)n * Kr + k] : 0.f;
  dst[idx] = (_Float16)v;
}

// weight fp32 [Kr,Nr] -> transposed f16 [Npad,Kpad]
__global__ void k_w16t(const float* __restrict__ src, _Float16* __restrict__ dst,
                       int Kr, int Nr, int Kpad, int Npad)
{
  size_t idx = (size_t)blockIdx.x * blockDim.x + threadIdx.x;
  if (idx >= (size_t)Npad * Kpad) return;
  int k = (int)(idx % Kpad), n = (int)(idx / Kpad);
  float v = (n < Nr && k < Kr) ? src[(size_t)k * Nr + n] : 0.f;
  dst[idx] = (_Float16)v;
}

// activation fp32 [Mr,Cr] -> f16 [Mpad,Kpad], zero pad
__global__ void k_act16(const float* __restrict__ src, _Float16* __restrict__ dst,
                        int Mr, int Cr, int Kpad, int Mpad)
{
  size_t idx = (size_t)blockIdx.x * blockDim.x + threadIdx.x;
  if (idx >= (size_t)Mpad * Kpad) return;
  int k = (int)(idx % Kpad);
  long m = (long)(idx / Kpad);
  float v = (m < Mr && k < Cr) ? src[m * Cr + k] : 0.f;
  dst[idx] = (_Float16)v;
}

// ---------------------------------------------------------------------------
// BatchNorm (training stats) over rows, per column; relu baked in (every BN in
// this net feeds a ReLU). Pre-BN biases omitted: BN(x+b) == BN(x).
// ---------------------------------------------------------------------------
__global__ void k_bn_stats(const float* __restrict__ X, float* __restrict__ mean,
                           float* __restrict__ var, int M, int C)
{
  __shared__ float ssum[256], ssq[256];
  int c = blockIdx.x;
  float s = 0.f, q = 0.f;
  for (int m = threadIdx.x; m < M; m += 256) {
    float v = X[(size_t)m * C + c];
    s += v; q += v * v;
  }
  ssum[threadIdx.x] = s; ssq[threadIdx.x] = q;
  __syncthreads();
  for (int o = 128; o > 0; o >>= 1) {
    if ((int)threadIdx.x < o) {
      ssum[threadIdx.x] += ssum[threadIdx.x + o];
      ssq[threadIdx.x]  += ssq[threadIdx.x + o];
    }
    __syncthreads();
  }
  if (threadIdx.x == 0) {
    float mu = ssum[0] / M;
    mean[c] = mu;
    var[c]  = ssq[0] / M - mu * mu;
  }
}

__global__ void k_bn_apply(const float* __restrict__ X, float* __restrict__ Y,
                           const float* __restrict__ mean, const float* __restrict__ var,
                           const float* __restrict__ g, const float* __restrict__ b,
                           int M, int C)
{
  size_t idx = (size_t)blockIdx.x * blockDim.x + threadIdx.x;
  if (idx >= (size_t)M * C) return;
  int c = (int)(idx % C);
  float v = (X[idx] - mean[c]) * rsqrtf(var[c] + BN_EPS) * g[c] + b[c];
  Y[idx] = fmaxf(v, 0.f);
}

// maxpool 2x2x2 stride 2, channels-last, -inf padding semantics
__global__ void k_maxpool(const float* __restrict__ src, float* __restrict__ dst,
                          int BT, int C, int Di, int Hi, int Wi,
                          int Do, int Ho, int Wo, int pd, int ph, int pw)
{
  size_t idx = (size_t)blockIdx.x * blockDim.x + threadIdx.x;
  size_t total = (size_t)BT * Do * Ho * Wo * C;
  if (idx >= total) return;
  int c = (int)(idx % C);
  long m = (long)(idx / C);
  int wo = (int)(m % Wo);
  int ho = (int)((m / Wo) % Ho);
  int dd = (int)((m / ((long)Wo * Ho)) % Do);
  int bt = (int)(m / ((long)Wo * Ho * Do));
  float best = -3.0e38f;
  for (int kd = 0; kd < 2; ++kd)
    for (int kh = 0; kh < 2; ++kh)
      for (int kw = 0; kw < 2; ++kw) {
        int d = 2 * dd - pd + kd, h = 2 * ho - ph + kh, w = 2 * wo - pw + kw;
        if (d < 0 || d >= Di || h < 0 || h >= Hi || w < 0 || w >= Wi) continue;
        float v = src[((((long)bt * Di + d) * Hi + h) * (long)Wi + w) * C + c];
        best = fmaxf(best, v);
      }
  dst[idx] = best;
}

// pool2 out [BT*8, 256] ch-last -> FC input f16 [Mpad, 2048] (torch NCDHW flatten)
__global__ void k_fc_rearr(const float* __restrict__ src, _Float16* __restrict__ dst,
                           int Mpad)
{
  size_t idx = (size_t)blockIdx.x * blockDim.x + threadIdx.x;
  if (idx >= (size_t)Mpad * 2048) return;
  int k = (int)(idx % 2048);
  int m = (int)(idx / 2048);
  float v = 0.f;
  if (m < 100) {
    int c = k >> 3, s = k & 7;
    v = src[((size_t)m * 8 + s) * 256 + c];
  }
  dst[idx] = (_Float16)v;
}

// GRU recurrence. Gi [Mpad,768] (rows bt=b*25+t) precomputed x@Wih^T (no bias).
// One workgroup: 1024 threads = (b,j) pairs, state h in LDS.
__global__ void __launch_bounds__(1024)
k_gru_scan(const float* __restrict__ Gi, const float* __restrict__ Whh,
           const float* __restrict__ bih, const float* __restrict__ bhh,
           float* __restrict__ Y)
{
  __shared__ float h[4 * 256];
  const int tid = threadIdx.x;
  const int b = tid >> 8, j = tid & 255;
  h[tid] = 0.f;
  __syncthreads();
  const float* wr = Whh + (size_t)j * 256;
  const float* wz = Whh + (size_t)(256 + j) * 256;
  const float* wn = Whh + (size_t)(512 + j) * 256;
  for (int t = 0; t < 25; ++t) {
    const float* gi = Gi + (size_t)(b * 25 + t) * 768;
    const float* hb = h + b * 256;
    float hr = bhh[j], hz = bhh[256 + j], hn = bhh[512 + j];
    for (int k = 0; k < 256; ++k) {
      float hk = hb[k];
      hr += wr[k] * hk; hz += wz[k] * hk; hn += wn[k] * hk;
    }
    float ir  = gi[j]       + bih[j];
    float iz  = gi[256 + j] + bih[256 + j];
    float in_ = gi[512 + j] + bih[512 + j];
    float rg = 1.f / (1.f + expf(-(ir + hr)));
    float zg = 1.f / (1.f + expf(-(iz + hz)));
    float ng = tanhf(in_ + rg * hn);
    float hnew = (1.f - zg) * ng + zg * hb[j];
    __syncthreads();
    h[tid] = hnew;
    Y[(size_t)(b * 25 + t) * 256 + j] = hnew;
    __syncthreads();
  }
}

// GRU output [100,256] (rows b*25+t) -> decoder input f16 [Mpad,6400]
__global__ void k_dec_rearr(const float* __restrict__ src, _Float16* __restrict__ dst,
                            int Mpad)
{
  size_t idx = (size_t)blockIdx.x * blockDim.x + threadIdx.x;
  if (idx >= (size_t)Mpad * 6400) return;
  int k = (int)(idx % 6400);
  int m = (int)(idx / 6400);
  float v = 0.f;
  if (m < 4) {
    int t = k >> 8, j = k & 255;
    v = src[((size_t)m * 25 + t) * 256 + j];
  }
  dst[idx] = (_Float16)v;
}

// node features f16 [N,160]: [:3]=node_x, [3:131]=enc[group], rest 0
__global__ void k_feats(const float* __restrict__ nodex, const float* __restrict__ enc,
                        _Float16* __restrict__ dst, int N, int npg)
{
  size_t idx = (size_t)blockIdx.x * blockDim.x + threadIdx.x;
  if (idx >= (size_t)N * 160) return;
  int k = (int)(idx % 160);
  int n = (int)(idx / 160);
  float v = 0.f;
  if (k < 3)        v = nodex[(size_t)n * 3 + k];
  else if (k < 131) v = enc[(size_t)(n / npg) * 128 + (k - 3)];
  dst[idx] = (_Float16)v;
}

__global__ void k_build_edges(const int* __restrict__ es, const int* __restrict__ ed,
                              int* __restrict__ S, int* __restrict__ D, int E0, int N)
{
  int i = blockIdx.x * blockDim.x + threadIdx.x;
  if (i >= E0 + N) return;
  if (i < E0) { S[i] = es[i]; D[i] = ed[i]; }
  else        { S[i] = i - E0; D[i] = i - E0; }
}

__global__ void k_fill(float* __restrict__ p, float v, size_t n)
{
  size_t i = (size_t)blockIdx.x * blockDim.x + threadIdx.x;
  if (i < n) p[i] = v;
}

__global__ void k_count(const int* __restrict__ D, float* __restrict__ CNT, int nE)
{
  int i = blockIdx.x * blockDim.x + threadIdx.x;
  if (i < nE) atomicAdd(&CNT[D[i]], 1.f);
}

__global__ void k_attn(const float* __restrict__ H, const float* __restrict__ asrc,
                       const float* __restrict__ adst, float* __restrict__ AS,
                       float* __restrict__ AD, int N, int F)
{
  int n = blockIdx.x * blockDim.x + threadIdx.x;
  if (n >= N) return;
  const float* h = H + (size_t)n * F;
  float s = 0.f, d = 0.f;
  for (int f = 0; f < F; ++f) { s += h[f] * asrc[f]; d += h[f] * adst[f]; }
  AS[n] = s; AD[n] = d;
}

static __device__ __forceinline__ void atomicMaxF(float* addr, float val)
{
  int* ia = (int*)addr;
  int old = __float_as_int(*addr);
  while (__int_as_float(old) < val) {
    int prev = atomicCAS(ia, old, __float_as_int(val));
    if (prev == old) break;
    old = prev;
  }
}

__global__ void k_elog(const int* __restrict__ S, const int* __restrict__ D,
                       const float* __restrict__ AS, const float* __restrict__ AD,
                       float* __restrict__ EB, float* __restrict__ DMAX, int nE)
{
  int e = blockIdx.x * blockDim.x + threadIdx.x;
  if (e >= nE) return;
  float v = AS[S[e]] + AD[D[e]];
  v = v > 0.f ? v : NEG_SLOPE * v;
  EB[e] = v;
  atomicMaxF(&DMAX[D[e]], v);
}

__global__ void k_eexp(const int* __restrict__ D, float* __restrict__ EB,
                       const float* __restrict__ DMAX, float* __restrict__ DSUM, int nE)
{
  int e = blockIdx.x * blockDim.x + threadIdx.x;
  if (e >= nE) return;
  float w = expf(EB[e] - DMAX[D[e]]);
  EB[e] = w;
  atomicAdd(&DSUM[D[e]], w);
}

// one block (F threads) per edge: agg[dst] += alpha * h[src]
__global__ void k_emsg(const int* __restrict__ S, const int* __restrict__ D,
                       const float* __restrict__ EB, const float* __restrict__ DSUM,
                       const float* __restrict__ H, float* __restrict__ AGG, int F)
{
  int e = blockIdx.x;
  int f = threadIdx.x;
  int d = D[e], s = S[e];
  float alpha = EB[e] / DSUM[d];
  atomicAdd(&AGG[(size_t)d * F + f], alpha * H[(size_t)s * F + f]);
}

__global__ void k_gfin(const float* __restrict__ AGG, const float* __restrict__ CNT,
                       const float* __restrict__ bias, float* __restrict__ X,
                       int N, int F)
{
  size_t idx = (size_t)blockIdx.x * blockDim.x + threadIdx.x;
  if (idx >= (size_t)N * F) return;
  int n = (int)(idx / F), f = (int)(idx % F);
  float v = AGG[idx] / CNT[n] + bias[f];
  X[idx] = fmaxf(v, 0.f);
}

// last GAT layer: F=1 projection + attention scalars
__global__ void k_finh(const float* __restrict__ X, const float* __restrict__ W,
                       const float* __restrict__ asrc, const float* __restrict__ adst,
                       float* __restrict__ H1, float* __restrict__ AS,
                       float* __restrict__ AD, int N, int K)
{
  int n = blockIdx.x * blockDim.x + threadIdx.x;
  if (n >= N) return;
  const float* x = X + (size_t)n * K;
  float s = 0.f;
  for (int k = 0; k < K; ++k) s += x[k] * W[k];
  H1[n] = s; AS[n] = s * asrc[0]; AD[n] = s * adst[0];
}

__global__ void k_emsg1(const int* __restrict__ S, const int* __restrict__ D,
                        const float* __restrict__ EB, const float* __restrict__ DSUM,
                        const float* __restrict__ H1, float* __restrict__ AGG1, int nE)
{
  int e = blockIdx.x * blockDim.x + threadIdx.x;
  if (e >= nE) return;
  int d = D[e];
  atomicAdd(&AGG1[d], (EB[e] / DSUM[d]) * H1[S[e]]);
}

__global__ void k_fout(const float* __restrict__ AGG1, const float* __restrict__ CNT,
                       const float* __restrict__ b, const float* __restrict__ y,
                       float* __restrict__ out, int N)
{
  int n = blockIdx.x * blockDim.x + threadIdx.x;
  if (n >= N) return;
  float v = AGG1[n] / CNT[n] + b[0];
  out[n]     = 1.f / (1.f + expf(-v));
  out[N + n] = y[n];
}

// ---------------------------------------------------------------------------
extern "C" void kernel_launch(void* const* d_in, const int* in_sizes, int n_in,
                              void* d_out, int out_size, void* d_ws, size_t ws_size,
                              hipStream_t stream)
{
  (void)n_in; (void)out_size; (void)ws_size;
  const float* Xb    = (const float*)d_in[0];
  const float* nodex = (const float*)d_in[1];
  const float* yin   = (const float*)d_in[2];
  const int*   esrc  = (const int*)d_in[3];
  const int*   edst  = (const int*)d_in[4];
  const int N   = in_sizes[2];          // 20000 (multiple of 32)
  const int E0  = in_sizes[3];          // 320000
  const int nE  = E0 + N;               // + self loops
  const int npg = N / 4;
  const int MC  = 18016;                // conv rows 18000 padded to %32
  const int MF  = 128;                  // FC/GRU rows 100 padded to %32
  const int MD  = 32;                   // decoder rows 4 padded to %32

  // ---- workspace carve (bump allocator, deterministic) ----
  char* ws = (char*)d_ws;
  size_t off = 0;
  auto carve = [&](size_t bytes) -> char* {
    char* p = ws + off; off += (bytes + 255) & ~(size_t)255; return p;
  };
  _Float16* COL = (_Float16*)carve((size_t)MC * 1728 * 2);    // im2col / A f16
  _Float16* W16 = (_Float16*)carve((size_t)512 * 6400 * 2);   // weight f16
  float* Cb   = (float*)carve((size_t)N * 256 * 4);           // GEMM out
  float* Xa   = (float*)carve((size_t)N * 256 * 4);           // activations
  float* AGG  = (float*)carve((size_t)N * 256 * 4);
  float* MEAN = (float*)carve(4096);
  float* VAR  = (float*)carve(4096);
  float* AS   = (float*)carve((size_t)N * 4);
  float* AD   = (float*)carve((size_t)N * 4);
  float* EB   = (float*)carve((size_t)nE * 4);
  float* DMAX = (float*)carve((size_t)N * 4);
  float* DSUM = (float*)carve((size_t)N * 4);
  float* CNT  = (float*)carve((size_t)N * 4);
  int*   ES   = (int*)carve((size_t)nE * 4);
  int*   ED   = (int*)carve((size_t)nE * 4);
  float* ENC  = (float*)carve(4 * 128 * 4);
  float* H1   = (float*)carve((size_t)N * 4);
  float* AG1  = (float*)carve((size_t)N * 4);

  const int TPB = 256;
  auto G1 = [](size_t n) { return dim3((unsigned)((n + 255) / 256)); };
  auto gemm = [&](const _Float16* A, const _Float16* B, float* Cc,
                  int Mpad, int Kpad, int Npad) {
    k_gemm_wmma<<<dim3((unsigned)(Npad / 64), (unsigned)(Mpad / 32)),
                  dim3(32), 0, stream>>>(A, B, Cc, Kpad, Npad);
  };
  auto bn = [&](const float* xin, float* xout, const void* g, const void* b,
                int M, int C) {
    k_bn_stats<<<dim3((unsigned)C), dim3(256), 0, stream>>>(xin, MEAN, VAR, M, C);
    k_bn_apply<<<G1((size_t)M * C), TPB, 0, stream>>>(
        xin, xout, MEAN, VAR, (const float*)g, (const float*)b, M, C);
  };

  // ================= CNN3D encoder (im2col + WMMA GEMM) =================
  // conv1: [100,5,5,6,6] -> [18000,64]
  k_im2col<<<G1((size_t)MC * 160), TPB, 0, stream>>>(Xb, COL, 100, 5, 5, 6, 6, 135, 160, MC, 1);
  k_w16<<<G1((size_t)64 * 160), TPB, 0, stream>>>((const float*)d_in[5], W16, 64, 135, 160, 64);
  gemm(COL, W16, Cb, MC, 160, 64);
  bn(Cb, Xa, d_in[7], d_in[8], 18000, 64);
  // conv2
  k_im2col<<<G1((size_t)MC * 1728), TPB, 0, stream>>>(Xa, COL, 100, 64, 5, 6, 6, 1728, 1728, MC, 0);
  k_w16<<<G1((size_t)64 * 1728), TPB, 0, stream>>>((const float*)d_in[9], W16, 64, 1728, 1728, 64);
  gemm(COL, W16, Cb, MC, 1728, 64);
  bn(Cb, Xa, d_in[11], d_in[12], 18000, 64);
  // maxpool(1,1,1): [5,6,6] -> [3,4,4]
  k_maxpool<<<G1((size_t)4800 * 64), TPB, 0, stream>>>(Xa, Cb, 100, 64, 5, 6, 6, 3, 4, 4, 1, 1, 1);
  // conv3: [100,64,3,4,4] -> [4800,256]
  k_im2col<<<G1((size_t)4800 * 1728), TPB, 0, stream>>>(Cb, COL, 100, 64, 3, 4, 4, 1728, 1728, 4800, 0);
  k_w16<<<G1((size_t)256 * 1728), TPB, 0, stream>>>((const float*)d_in[13], W16, 256, 1728, 1728, 256);
  gemm(COL, W16, Cb, 4800, 1728, 256);
  bn(Cb, Xa, d_in[15], d_in[16], 4800, 256);
  // maxpool(1,0,0): [3,4,4] -> [2,2,2]
  k_maxpool<<<G1((size_t)800 * 256), TPB, 0, stream>>>(Xa, Cb, 100, 256, 3, 4, 4, 2, 2, 2, 1, 0, 0);

  // ================= FC encoder =================
  k_fc_rearr<<<G1((size_t)MF * 2048), TPB, 0, stream>>>(Cb, COL, MF);
  k_w16<<<G1((size_t)512 * 2048), TPB, 0, stream>>>((const float*)d_in[17], W16, 512, 2048, 2048, 512);
  gemm(COL, W16, Cb, MF, 2048, 512);
  bn(Cb, Xa, d_in[19], d_in[20], 100, 512);
  k_act16<<<G1((size_t)MF * 512), TPB, 0, stream>>>(Xa, COL, 100, 512, 512, MF);
  k_w16<<<G1((size_t)256 * 512), TPB, 0, stream>>>((const float*)d_in[21], W16, 256, 512, 512, 256);
  gemm(COL, W16, Cb, MF, 512, 256);
  bn(Cb, Xa, d_in[23], d_in[24], 100, 256);

  // ================= GRU (2 layers) =================
  for (int l = 0; l < 2; ++l) {
    int p = 25 + 4 * l;
    k_act16<<<G1((size_t)MF * 256), TPB, 0, stream>>>(Xa, COL, 100, 256, 256, MF);
    k_w16<<<G1((size_t)768 * 256), TPB, 0, stream>>>((const float*)d_in[p], W16, 768, 256, 256, 768);
    gemm(COL, W16, Cb, MF, 256, 768);   // Gi = x @ Wih^T
    k_gru_scan<<<dim3(1), dim3(1024), 0, stream>>>(
        Cb, (const float*)d_in[p + 1], (const float*)d_in[p + 2],
        (const float*)d_in[p + 3], Xa);
  }

  // ================= decoder =================
  k_dec_rearr<<<G1((size_t)MD * 6400), TPB, 0, stream>>>(Xa, COL, MD);
  k_w16<<<G1((size_t)512 * 6400), TPB, 0, stream>>>((const float*)d_in[33], W16, 512, 6400, 6400, 512);
  gemm(COL, W16, Cb, MD, 6400, 512);
  bn(Cb, Xa, d_in[35], d_in[36], 4, 512);
  k_act16<<<G1((size_t)MD * 512), TPB, 0, stream>>>(Xa, COL, 4, 512, 512, MD);
  k_w16<<<G1((size_t)128 * 512), TPB, 0, stream>>>((const float*)d_in[37], W16, 128, 512, 512, 128);
  gemm(COL, W16, Cb, MD, 512, 128);
  bn(Cb, ENC, d_in[39], d_in[40], 4, 128);

  // ================= graph prep =================
  k_build_edges<<<G1((size_t)nE), TPB, 0, stream>>>(esrc, edst, ES, ED, E0, N);
  k_fill<<<G1((size_t)N), TPB, 0, stream>>>(CNT, 0.f, (size_t)N);
  k_count<<<G1((size_t)nE), TPB, 0, stream>>>(ED, CNT, nE);
  k_feats<<<G1((size_t)N * 160), TPB, 0, stream>>>(nodex, ENC, COL, N, npg);

  // ================= 6x (GATConv + ReLU) =================
  for (int i = 0; i < 6; ++i) {
    const int Kr = i ? 256 : 131;
    const int Kp = i ? 256 : 160;
    if (i) k_act16<<<G1((size_t)N * 256), TPB, 0, stream>>>(Xa, COL, N, 256, 256, N);
    k_w16t<<<G1((size_t)256 * Kp), TPB, 0, stream>>>((const float*)d_in[41 + 4 * i], W16, Kr, 256, Kp, 256);
    gemm(COL, W16, Cb, N, Kp, 256);                       // h = x @ W
    k_attn<<<G1((size_t)N), TPB, 0, stream>>>(Cb, (const float*)d_in[42 + 4 * i],
                                              (const float*)d_in[43 + 4 * i], AS, AD, N, 256);
    k_fill<<<G1((size_t)N), TPB, 0, stream>>>(DMAX, -3.0e38f, (size_t)N);
    k_fill<<<G1((size_t)N), TPB, 0, stream>>>(DSUM, 0.f, (size_t)N);
    k_fill<<<G1((size_t)N * 256), TPB, 0, stream>>>(AGG, 0.f, (size_t)N * 256);
    k_elog<<<G1((size_t)nE), TPB, 0, stream>>>(ES, ED, AS, AD, EB, DMAX, nE);
    k_eexp<<<G1((size_t)nE), TPB, 0, stream>>>(ED, EB, DMAX, DSUM, nE);
    k_emsg<<<dim3((unsigned)nE), dim3(256), 0, stream>>>(ES, ED, EB, DSUM, Cb, AGG, 256);
    k_gfin<<<G1((size_t)N * 256), TPB, 0, stream>>>(AGG, CNT, (const float*)d_in[44 + 4 * i], Xa, N, 256);
  }

  // ================= final GATConv (F=1) + sigmoid =================
  k_finh<<<G1((size_t)N), TPB, 0, stream>>>(Xa, (const float*)d_in[65],
                                            (const float*)d_in[66], (const float*)d_in[67],
                                            H1, AS, AD, N, 256);
  k_fill<<<G1((size_t)N), TPB, 0, stream>>>(DMAX, -3.0e38f, (size_t)N);
  k_fill<<<G1((size_t)N), TPB, 0, stream>>>(DSUM, 0.f, (size_t)N);
  k_fill<<<G1((size_t)N), TPB, 0, stream>>>(AG1, 0.f, (size_t)N);
  k_elog<<<G1((size_t)nE), TPB, 0, stream>>>(ES, ED, AS, AD, EB, DMAX, nE);
  k_eexp<<<G1((size_t)nE), TPB, 0, stream>>>(ED, EB, DMAX, DSUM, nE);
  k_emsg1<<<G1((size_t)nE), TPB, 0, stream>>>(ES, ED, EB, DSUM, H1, AG1, nE);
  k_fout<<<G1((size_t)N), TPB, 0, stream>>>(AG1, CNT, (const float*)d_in[68], yin, (float*)d_out, N);
}
